// GraphLinConv_87419764343142
// MI455X (gfx1250) — matrix operand order repaired and checked by reading, not compiled, not run on hardware
//
#include <hip/hip_runtime.h>

typedef __attribute__((ext_vector_type(16))) _Float16 v16h;
typedef __attribute__((ext_vector_type(8)))  _Float16 v8h;
typedef __attribute__((ext_vector_type(8)))  float    v8f;

#define B_   8
#define T_   300
#define A_   3
#define V_   25
#define P_   2
#define C_   64
#define F_   64
#define N_   4
#define K_   9
#define TT_  292            // T - K + 1 (valid temporal length)
#define TB_  64             // time-block per workgroup
#define TL_  (TB_ + K_ - 1) // 72 staged time steps
#define NTB_ 5              // 5*64 = 320 >= 300: covers the zero-pad tail too

// ---- weight f32 -> f16 conversion (runs once per launch into d_ws) ----
__global__ void GraphLinConv_cvtw(const float* __restrict__ w,
                                  _Float16* __restrict__ o, int n) {
  int i = blockIdx.x * blockDim.x + threadIdx.x;
  if (i < n) o[i] = (_Float16)w[i];
}

// LDS layout: row = tl*8 + n*2 + p (576 rows of 64 halfs = 72KB), with the
// 8-half (16B) chunks XOR-swizzled by (tl & 7): A-fragment loads step tl by 1
// per lane (row stride 8*64 halfs == bank-aligned), so the swizzle spreads the
// 16 M-lanes across banks (residual 2-way conflict at m vs m+8 only).
__device__ __forceinline__ int lds_off(int tl, int n, int p, int c) {
  int row = (tl * N_ + n) * P_ + p;
  int chunk = ((c >> 3) ^ (tl & 7)) << 3;
  return row * C_ + chunk + (c & 7);
}

__global__ __launch_bounds__(256)
void GraphLinConv_87419764343142_kernel(const float* __restrict__ x,
                                        const _Float16* __restrict__ w,
                                        const int* __restrict__ nbr,
                                        float* __restrict__ out) {
  __shared__ __align__(16) _Float16 xs[TL_ * N_ * P_ * C_]; // 36864 halfs = 72KB

  const int blk = blockIdx.x;
  const int tb = blk % NTB_;
  const int u  = (blk / NTB_) % V_;
  const int a  = (blk / (NTB_ * V_)) % A_;
  const int b  = blk / (NTB_ * V_ * A_);
  const int t0 = tb * TB_;
  const int tid = threadIdx.x;

  // ---- stage gathered x tile into LDS as f16 ----
  // 576 rows x 64 ch; unit = 8 halfs (one swizzle chunk, one ds_store_b128):
  // 4608 units / 256 threads = 18 iterations
#pragma unroll 3
  for (int it = 0; it < (TL_ * N_ * P_ * C_ / 8) / 256; ++it) {
    int q   = it * 256 + tid;
    int c   = (q * 8) & (C_ - 1);
    int row = q >> 3;                 // q / (C_/8)
    int pp  = row & 1;
    int nn  = (row >> 1) & 3;
    int tl  = row >> 3;
    int tg  = t0 + tl;
    v8h h;
    if (tg < T_) {
      int v = nbr[u * N_ + nn];
      const float4* src = (const float4*)(x +
          ((((size_t)b * T_ + tg) * A_ + a) * (V_ * P_) + (v * P_ + pp)) * C_ + c);
      float4 f0v = src[0];
      float4 f1v = src[1];
      h[0] = (_Float16)f0v.x; h[1] = (_Float16)f0v.y;
      h[2] = (_Float16)f0v.z; h[3] = (_Float16)f0v.w;
      h[4] = (_Float16)f1v.x; h[5] = (_Float16)f1v.y;
      h[6] = (_Float16)f1v.z; h[7] = (_Float16)f1v.w;
    } else {
      h = (v8h)((_Float16)0.f);
    }
    *(v8h*)(&xs[lds_off(tl, nn, pp, c)]) = h;
  }
  __syncthreads();

  // ---- per-wave 32t x 32F tile via WMMA (4 accumulators) ----
  const int wave  = tid >> 5;
  const int lane  = tid & 31;
  const int tpair = wave & 1;        // which pair of 16-t subtiles
  const int p     = (wave >> 1) & 1; // person
  const int fh    = wave >> 2;       // F half (0 or 1)
  const int f0    = fh * 32;
  const int tbase = tpair * 32;      // local t of this wave's first M row

  const int m  = lane & 15;          // A: row M / B,D: column f
  const int hi = lane >> 4;          // lane-half selector

  v8f acc00 = {};  // (t-tile 0, f-tile 0)
  v8f acc01 = {};  // (t-tile 0, f-tile 1)
  v8f acc10 = {};  // (t-tile 1, f-tile 0)
  v8f acc11 = {};  // (t-tile 1, f-tile 1)

  for (int k = 0; k < K_; ++k) {
    const int tl0 = tbase + m + k;       // A rows, first 16-t tile
    const int tl1 = tl0 + 16;            // second 16-t tile
#pragma unroll
    for (int n = 0; n < N_; ++n) {
#pragma unroll
      for (int cc = 0; cc < 2; ++cc) {
        const int c0 = cc * 32;
        // A fragments (16x32 f16): elems 0-7 = c0+hi*16.., elems 8-15 = c0+hi*16+8..
        v8h a0lo = *(const v8h*)(&xs[lds_off(tl0, n, p, c0 + hi * 16)]);
        v8h a0hi = *(const v8h*)(&xs[lds_off(tl0, n, p, c0 + hi * 16 + 8)]);
        v8h a1lo = *(const v8h*)(&xs[lds_off(tl1, n, p, c0 + hi * 16)]);
        v8h a1hi = *(const v8h*)(&xs[lds_off(tl1, n, p, c0 + hi * 16 + 8)]);
        v16h av0 = __builtin_shufflevector(a0lo, a0hi,
            0, 1, 2, 3, 4, 5, 6, 7, 8, 9, 10, 11, 12, 13, 14, 15);
        v16h av1 = __builtin_shufflevector(a1lo, a1hi,
            0, 1, 2, 3, 4, 5, 6, 7, 8, 9, 10, 11, 12, 13, 14, 15);
        // B fragments (32x16 f16): lane = column f, elements = 16 contiguous c
        const _Float16* wb = w + ((size_t)(n * K_ + k) * F_) * C_ + c0 + hi * 16;
        v16h b0 = *(const v16h*)(wb + (size_t)(f0 + m) * C_);
        v16h b1 = *(const v16h*)(wb + (size_t)(f0 + 16 + m) * C_);
        // 4 WMMAs reuse both A and B fragments
        acc00 = __builtin_amdgcn_wmma_f32_16x16x32_f16(
            false, av0, false, b0, (short)0, acc00, false, false);
        acc01 = __builtin_amdgcn_wmma_f32_16x16x32_f16(
            false, av0, false, b1, (short)0, acc01, false, false);
        acc10 = __builtin_amdgcn_wmma_f32_16x16x32_f16(
            false, av1, false, b0, (short)0, acc10, false, false);
        acc11 = __builtin_amdgcn_wmma_f32_16x16x32_f16(
            false, av1, false, b1, (short)0, acc11, false, false);
      }
    }
  }

  // ---- store: C/D layout is VGPR r, lanes<16 -> M=r, lanes>=16 -> M=r+8 ----
  const size_t obase = (((size_t)b * T_) * A_ + a) * (V_ * P_) + (u * P_ + p);
#pragma unroll
  for (int r = 0; r < 8; ++r) {
    int trow = r + hi * 8;
    int ta = t0 + tbase + trow;        // t-tile 0
    int tb2 = ta + 16;                 // t-tile 1
    if (ta < T_) {
      size_t o = (obase + (size_t)ta * A_ * (V_ * P_)) * F_ + f0 + m;
      out[o]      = (ta < TT_) ? acc00[r] : 0.0f;  // zero-pad tail of time
      out[o + 16] = (ta < TT_) ? acc01[r] : 0.0f;
    }
    if (tb2 < T_) {
      size_t o = (obase + (size_t)tb2 * A_ * (V_ * P_)) * F_ + f0 + m;
      out[o]      = (tb2 < TT_) ? acc10[r] : 0.0f;
      out[o + 16] = (tb2 < TT_) ? acc11[r] : 0.0f;
    }
  }
}

extern "C" void kernel_launch(void* const* d_in, const int* in_sizes, int n_in,
                              void* d_out, int out_size, void* d_ws, size_t ws_size,
                              hipStream_t stream) {
  const float* x    = (const float*)d_in[0];
  const float* kern = (const float*)d_in[1];
  const int*   nbr  = (const int*)d_in[2];
  float*       out  = (float*)d_out;
  _Float16*    wf16 = (_Float16*)d_ws;  // 4*9*64*64 f16 = 294912 bytes

  const int nw = N_ * K_ * F_ * C_;
  GraphLinConv_cvtw<<<(nw + 255) / 256, 256, 0, stream>>>(kern, wf16, nw);

  const int nblk = B_ * A_ * V_ * NTB_;  // 3000 workgroups
  GraphLinConv_87419764343142_kernel<<<nblk, 256, 0, stream>>>(x, wf16, nbr, out);
}
